// ChannelMerger_5626407158063
// MI455X (gfx1250) — compile-verified, matrix operand order checked by reading, not run on hardware
//
#include <hip/hip_runtime.h>
#include <hip/hip_bf16.h>
#include <math.h>

// Sizes (fixed by the reference)
#define B_  32
#define C_  273
#define T_  4096
#define O_  270
#define D_  288      // POS_DIM, = 9 * 32 exactly
#define OPAD 272     // 17 tiles of 16
#define CPAD 288     // 18 tiles of 16 / 9 k-steps of 32
#define NKO 9        // K-steps of 32 over D=288
#define NKC 9        // K-steps of 32 over CPAD=288
#define NOT 17       // o tiles
#define NCT 18       // c tiles

// wfrag slab per batch: NOT * NKC * 512 bf16 elements
#define WFRAG_ELEMS_PER_B (NOT * NKC * 512)            // 78336
#define WFRAG_BYTES_PER_B (WFRAG_ELEMS_PER_B * 2)      // 156672
#define WFRAG_QW_PER_B    (WFRAG_BYTES_PER_B / 8)      // 19584 8-byte elements

typedef __attribute__((ext_vector_type(16))) __bf16 v16bf;
typedef __attribute__((ext_vector_type(8)))  __bf16 v8bf;
typedef __attribute__((ext_vector_type(8)))  float  v8f;
typedef __attribute__((ext_vector_type(4)))  unsigned int v4u;
typedef __attribute__((ext_vector_type(8)))  int v8i;
typedef __attribute__((ext_vector_type(4)))  int v4i;

// ---------------------------------------------------------------------------
// K0a: fourier embedding -> bf16, padded [B][CPAD][D]
// ---------------------------------------------------------------------------
__global__ void emb_kernel(const float* __restrict__ pos, __bf16* __restrict__ emb) {
  int c = blockIdx.x;            // 0..272
  int b = blockIdx.y;            // 0..31
  int d = threadIdx.x;           // 0..287
  float px = pos[((size_t)b * C_ + c) * 2 + 0] + 0.1f;
  float py = pos[((size_t)b * C_ + c) * 2 + 1] + 0.1f;
  int idx = (d < 144) ? d : d - 144;
  int fx = idx / 12, fy = idx - fx * 12;
  const float scale = 6.283185307179586f / 1.2f;   // 2*pi / (1 + 2*margin)
  float loc = scale * (px * (float)fx + py * (float)fy);
  float s, cc;
  __sincosf(loc, &s, &cc);
  float v = (d < 144) ? cc : s;
  emb[((size_t)b * CPAD + c) * D_ + d] = (__bf16)v;
}

// ---------------------------------------------------------------------------
// K0b: heads -> bf16, padded [OPAD][D], rows >= 270 zeroed
// ---------------------------------------------------------------------------
__global__ void pad_heads_kernel(const float* __restrict__ heads, __bf16* __restrict__ hb) {
  int r = blockIdx.x;            // 0..271
  int d = threadIdx.x;           // 0..287
  float v = (r < O_) ? heads[(size_t)r * D_ + d] : 0.0f;
  hb[(size_t)r * D_ + d] = (__bf16)v;
}

// WMMA 16-bit fragment K index for element j of the per-lane v16bf
// lane<16: K = {0..7, 16..23};  lane>=16: K = {8..15, 24..31}
__device__ __forceinline__ int frag_k(int j, int half) {
  return (j < 8) ? (j + (half ? 8 : 0)) : (j + (half ? 16 : 8));
}

// ---------------------------------------------------------------------------
// K1: scores[b][OPAD][CPAD] (f32) = heads(bf16) x emb^T  via WMMA bf16
// ---------------------------------------------------------------------------
__global__ void __launch_bounds__(32) score_kernel(const __bf16* __restrict__ hb,
                                                   const __bf16* __restrict__ emb,
                                                   float* __restrict__ scores) {
  int ct = blockIdx.x;           // 0..17
  int ot = blockIdx.y;           // 0..16
  int b  = blockIdx.z;           // 0..31
  int lane = threadIdx.x;        // 0..31 (wave32)
  int half = lane >> 4;
  int m    = lane & 15;
  const __bf16* hrow = hb + (size_t)(ot * 16 + m) * D_;               // padded, valid
  const __bf16* erow = emb + ((size_t)b * CPAD + (ct * 16 + m)) * D_; // padded, valid

  v8f acc = {};
#pragma unroll
  for (int k = 0; k < NKO; ++k) {
    int d0 = 32 * k + (half ? 8 : 0);     // j = 0..7  -> d0 + j
    int d1 = 32 * k + (half ? 24 : 16);   // j = 8..15 -> d1 + (j-8)
    v8bf alo = *(const v8bf*)(hrow + d0);
    v8bf ahi = *(const v8bf*)(hrow + d1);
    v8bf blo = *(const v8bf*)(erow + d0);
    v8bf bhi = *(const v8bf*)(erow + d1);
    v16bf a, bm;
#pragma unroll
    for (int j = 0; j < 8; ++j) {
      a[j] = alo[j];  a[j + 8] = ahi[j];
      bm[j] = blo[j]; bm[j + 8] = bhi[j];
    }
    acc = __builtin_amdgcn_wmma_f32_16x16x32_bf16(false, a, false, bm,
                                                  (short)0, acc, false, false);
  }
  float* srow = scores + (size_t)b * OPAD * CPAD;
  int col = ct * 16 + (lane & 15);
#pragma unroll
  for (int r = 0; r < 8; ++r) {
    int row = ot * 16 + r + (half ? 8 : 0);      // < OPAD always
    srow[(size_t)row * CPAD + col] = acc[r];
  }
}

// ---------------------------------------------------------------------------
// K2: softmax over C per (b,o) row (one wave each), emit weights as bf16
// pre-swizzled WMMA A-fragments: wfrag[b][ot][k][flane][16], zero for c>=273
// ---------------------------------------------------------------------------
__global__ void __launch_bounds__(256) softmax_pack_kernel(const float* __restrict__ scores,
                                                           const unsigned char* __restrict__ mask,
                                                           __bf16* __restrict__ wfrag) {
  int wid  = (blockIdx.x * blockDim.x + threadIdx.x) >> 5;   // 0..8639
  int lane = threadIdx.x & 31;
  if (wid >= B_ * O_) return;
  int b = wid / O_, o = wid - b * O_;
  const float* srow = scores + ((size_t)b * OPAD + o) * CPAD;

  float sv[9];
  float mx = -3.0e38f;
#pragma unroll
  for (int it = 0; it < 9; ++it) {
    int c = lane + 32 * it;
    float v = -3.0e38f;
    if (c < C_) {
      v = srow[c];
      if (mask[(size_t)b * C_ + c]) v = -3.0e38f;   // invalid channel -> -inf
    }
    sv[it] = v;
    mx = fmaxf(mx, v);
  }
#pragma unroll
  for (int off = 16; off > 0; off >>= 1) mx = fmaxf(mx, __shfl_xor(mx, off, 32));
  float sum = 0.0f;
#pragma unroll
  for (int it = 0; it < 9; ++it) {
    int c = lane + 32 * it;
    float e = (c < C_) ? __expf(sv[it] - mx) : 0.0f;
    sv[it] = e;
    sum += e;
  }
#pragma unroll
  for (int off = 16; off > 0; off >>= 1) sum += __shfl_xor(sum, off, 32);
  float inv = 1.0f / sum;

  int ot = o >> 4, m = o & 15;
  __bf16* base = wfrag + ((size_t)b * NOT + ot) * (size_t)NKC * 512;
#pragma unroll
  for (int it = 0; it < 9; ++it) {
    int c = lane + 32 * it;                     // < CPAD
    float w = (c < C_) ? sv[it] * inv : 0.0f;   // explicit zero K-padding
    int k = c >> 5, kk = c & 31;
    int flane = m + ((kk & 8) ? 16 : 0);
    int j = (kk & 7) + ((kk & 16) ? 8 : 0);
    base[(size_t)k * 512 + flane * 16 + j] = (__bf16)w;
  }
}

// ---------------------------------------------------------------------------
// K3: out[b][O][T] = weights[b][O][C] x meg[b][C][T]  via WMMA bf16
// Workgroup = 8 waves = one (b, 128-column t-slab).
//  - The batch's entire weight-fragment slab (156,672 B) is DMA'd into LDS by
//    the Tensor Data Mover (one tensor_load_to_lds, TENSORcnt-tracked) while
//    all 8 waves stream their meg B-fragments from HBM -> staging latency is
//    fully hidden behind the dominant global traffic.
//  - Each wave keeps its 9 meg B-fragments in VGPRs (meg read exactly once,
//    nontemporal) and loops all 17 o-tiles against LDS A-fragments.
// ---------------------------------------------------------------------------
__global__ void __launch_bounds__(256) merge_gemm_kernel(const float* __restrict__ meg,
                                                         const __bf16* __restrict__ wfrag,
                                                         float* __restrict__ out) {
  extern __shared__ __bf16 smem[];             // WFRAG_BYTES_PER_B bytes
  int b    = blockIdx.y;
  int tid  = threadIdx.x;
  int lane = tid & 31;
  int wave = tid >> 5;
  int t0   = blockIdx.x * 128 + wave * 16;
  int n    = lane & 15;
  int half = lane >> 4;

  // --- TDM: DMA this batch's weight-fragment slab into LDS (wave 0 only) ---
  if (wave == 0) {
    unsigned long long ga = (unsigned long long)(wfrag + (size_t)b * WFRAG_ELEMS_PER_B);
    unsigned int lds_addr = (unsigned int)(unsigned long long)(uintptr_t)smem; // LDS aperture low bits
    // D# group 0: count=1 | lds_addr | global_addr[56:0] | type=2
    v4u g0;
    g0[0] = 1u;                                         // count = 1 (valid user D#)
    g0[1] = lds_addr;                                   // lds_addr (bytes)
    g0[2] = (unsigned int)(ga & 0xFFFFFFFFull);         // global_addr[31:0]
    g0[3] = (unsigned int)((ga >> 32) & 0x1FFFFFFull) | (2u << 30); // addr[56:32] | type=2
    // D# group 1: data_size=8B; tensor_dim0 = tile_dim0 = 19584; dim1 = 1 line
    v8i g1;
    g1[0] = (int)(3u << 16);                            // workgroup_mask=0, data_size=3 (8B)
    g1[1] = (int)((WFRAG_QW_PER_B & 0xFFFF) << 16);     // tensor_dim0[15:0] @ bits 63:48
    g1[2] = (int)(((WFRAG_QW_PER_B >> 16) & 0xFFFF) | (1u << 16)); // dim0[31:16] | tensor_dim1=1
    g1[3] = (int)((WFRAG_QW_PER_B & 0xFFFF) << 16);     // tensor_dim1[31:16]=0 | tile_dim0 @127:112
    g1[4] = 1;                                          // tile_dim1 = 1, tile_dim2 = 0
    g1[5] = (int)WFRAG_QW_PER_B;                        // tensor_dim0_stride[31:0]
    g1[6] = 0;                                          // stride hi | tensor_dim1_stride lo
    g1[7] = 0;
    v4i gz4 = {0, 0, 0, 0};
    v8i gz8 = {0, 0, 0, 0, 0, 0, 0, 0};
    __builtin_amdgcn_tensor_load_to_lds(g0, g1, gz4, gz4, gz8, 0);
  }

  // Meanwhile build the 9 meg B-fragments in VGPRs (independent of the DMA)
  const float* megb = meg + (size_t)b * C_ * T_ + (t0 + n);
  v16bf bfrag[NKC];
#pragma unroll
  for (int k = 0; k < NKC; ++k) {
#pragma unroll
    for (int j = 0; j < 16; ++j) {
      int c = 32 * k + frag_k(j, half);
      float v = (c < C_) ? __builtin_nontemporal_load(megb + (size_t)c * T_) : 0.0f;
      bfrag[k][j] = (__bf16)v;
    }
  }

  if (wave == 0) __builtin_amdgcn_s_wait_tensorcnt(0);  // DMA complete
  __syncthreads();                                      // publish LDS to all waves

  const __bf16* wb = smem + (size_t)lane * 16;
  float* outb = out + (size_t)b * O_ * T_ + t0 + n;
  for (int ot = 0; ot < NOT; ++ot) {
    v8f acc = {};
#pragma unroll
    for (int k = 0; k < NKC; ++k) {
      const __bf16* ap = wb + ((size_t)ot * NKC + k) * 512;
      v8bf lo = *(const v8bf*)(ap);       // ds_load_b128
      v8bf hi = *(const v8bf*)(ap + 8);   // ds_load_b128
      v16bf a;
#pragma unroll
      for (int j = 0; j < 8; ++j) { a[j] = lo[j]; a[j + 8] = hi[j]; }
      acc = __builtin_amdgcn_wmma_f32_16x16x32_bf16(false, a, false, bfrag[k],
                                                    (short)0, acc, false, false);
    }
#pragma unroll
    for (int r = 0; r < 8; ++r) {
      int row = ot * 16 + r + (half ? 8 : 0);
      if (row < O_)
        __builtin_nontemporal_store(acc[r], outb + (size_t)row * T_);
    }
  }
}

// ---------------------------------------------------------------------------
extern "C" void kernel_launch(void* const* d_in, const int* in_sizes, int n_in,
                              void* d_out, int out_size, void* d_ws, size_t ws_size,
                              hipStream_t stream) {
  const float* meg   = (const float*)d_in[0];           // [B,C,T]
  const float* pos   = (const float*)d_in[1];           // [B,C,2]
  const float* heads = (const float*)d_in[2];           // [O,D]
  const unsigned char* mask = (const unsigned char*)d_in[3]; // [B,C] bool
  float* out = (float*)d_out;                           // [B,O,T]

  // Workspace layout (all 256-byte aligned)
  char* ws = (char*)d_ws;
  size_t emb_bytes    = (size_t)B_ * CPAD * D_ * sizeof(__bf16);       // ~5.3 MB
  size_t scores_bytes = (size_t)B_ * OPAD * CPAD * sizeof(float);      // ~10 MB
  size_t wfrag_bytes  = (size_t)B_ * WFRAG_BYTES_PER_B;                // ~5.0 MB
  __bf16* emb_ws    = (__bf16*)ws;
  float*  scores_ws = (float*)(ws + ((emb_bytes + 255) & ~(size_t)255));
  __bf16* wfrag_ws  = (__bf16*)((char*)scores_ws + ((scores_bytes + 255) & ~(size_t)255));
  __bf16* hb_ws     = (__bf16*)((char*)wfrag_ws + ((wfrag_bytes + 255) & ~(size_t)255));

  // K0: embeddings + padded bf16 heads (tiny)
  emb_kernel<<<dim3(C_, B_), D_, 0, stream>>>(pos, emb_ws);
  pad_heads_kernel<<<OPAD, D_, 0, stream>>>(heads, hb_ws);
  // K1: scores via WMMA (branch-free fragments)
  score_kernel<<<dim3(NCT, NOT, B_), 32, 0, stream>>>(hb_ws, emb_ws, scores_ws);
  // K2: softmax + A-fragment pack (8640 waves)
  int nwaves = B_ * O_;
  softmax_pack_kernel<<<(nwaves + 7) / 8, 256, 0, stream>>>(scores_ws, mask, wfrag_ws);
  // K3: main GEMM (memory-bound, ~285 MB HBM traffic; A-frags via TDM -> LDS)
  merge_gemm_kernel<<<dim3(T_ / 128, B_), 256, WFRAG_BYTES_PER_B, stream>>>(meg, wfrag_ws, out);
}
